// VectorQuantizer_56444460204638
// MI455X (gfx1250) — matrix operand order repaired and checked by reading, not compiled, not run on hardware
//
#include <hip/hip_runtime.h>
#include <hip/hip_bf16.h>

// ---------------------------------------------------------------------------
// VQ (VectorQuantizer) forward for MI455X / gfx1250, wave32 + WMMA bf16.
//   z:        [16, 256, 32, 32] f32   (B,C,H,W)
//   codebook: [8192, 256] f32
// Outputs (concatenated in d_out):
//   z_q channel-first [16,256,32,32] f32  (4194304)
//   index [16,32,32] int32                (16384, raw bits)
//   loss f32                              (1)
// ---------------------------------------------------------------------------

#define NPOS   16384      // 16 * 32 * 32 positions
#define KDIM   256        // channels
#define NE     8192       // codebook entries
#define HW     1024       // 32*32
#define NSPLIT 16         // N-dimension split across blocks
#define NT_PER 32         // 16-wide N tiles per split (512/16)
#define BROW   528        // padded LDS row stride in bytes (256*2 + 16)
#define BTILE  (16 * BROW) // 8448 bytes per staged B tile

typedef __bf16 v16bf __attribute__((ext_vector_type(16)));
typedef float  v8f   __attribute__((ext_vector_type(8)));

__device__ __forceinline__ unsigned short f32_to_bf16_rne(float f) {
    unsigned int u = __float_as_uint(f);
    u += 0x7FFFu + ((u >> 16) & 1u);   // round to nearest even
    return (unsigned short)(u >> 16);
}

// CDNA5 async copy: global -> LDS, 16B per lane, tracked by ASYNCcnt.
// ldsaddr: LDS byte address (low 32 bits of generic pointer to __shared__).
__device__ __forceinline__ void async_copy_b128(unsigned int ldsaddr, const void* g) {
    asm volatile("global_load_async_to_lds_b128 %0, %1, off"
                 :: "v"(ldsaddr), "v"((unsigned long long)g)
                 : "memory");
}

__device__ __forceinline__ void wait_asynccnt0() {
    asm volatile("s_wait_asynccnt 0" ::: "memory");
}

// ---------------------------------------------------------------------------
// Prep 1: codebook f32 -> bf16, plus per-row squared norm.
// One wave per codebook row; 8 waves (256 threads) per block; 1024 blocks.
// ---------------------------------------------------------------------------
__global__ __launch_bounds__(256) void vq_prep_codebook(
    const float* __restrict__ cb_f32,
    unsigned short* __restrict__ cb_bf16,
    float* __restrict__ cnorm)
{
    const int wave = threadIdx.x >> 5;
    const int lane = threadIdx.x & 31;
    const int row  = blockIdx.x * 8 + wave;          // 0..8191

    const float4* src = (const float4*)(cb_f32 + row * KDIM + lane * 8);
    float4 f0 = src[0];
    float4 f1 = src[1];

    float s = f0.x*f0.x + f0.y*f0.y + f0.z*f0.z + f0.w*f0.w
            + f1.x*f1.x + f1.y*f1.y + f1.z*f1.z + f1.w*f1.w;

    uint4 pk;
    pk.x = (unsigned int)f32_to_bf16_rne(f0.x) | ((unsigned int)f32_to_bf16_rne(f0.y) << 16);
    pk.y = (unsigned int)f32_to_bf16_rne(f0.z) | ((unsigned int)f32_to_bf16_rne(f0.w) << 16);
    pk.z = (unsigned int)f32_to_bf16_rne(f1.x) | ((unsigned int)f32_to_bf16_rne(f1.y) << 16);
    pk.w = (unsigned int)f32_to_bf16_rne(f1.z) | ((unsigned int)f32_to_bf16_rne(f1.w) << 16);
    *(uint4*)(cb_bf16 + row * KDIM + lane * 8) = pk;

    #pragma unroll
    for (int off = 16; off > 0; off >>= 1)
        s += __shfl_xor(s, off, 32);
    if (lane == 0) cnorm[row] = s;
}

// ---------------------------------------------------------------------------
// Prep 2: z [B,C,H,W] f32 -> z_cl [pos, C] bf16 via LDS 32x32 tile transpose.
// grid = (32 hw-tiles, 8 c-tiles, 16 batches); coalesced on both sides.
// ---------------------------------------------------------------------------
__global__ __launch_bounds__(256) void vq_prep_z(
    const float* __restrict__ z,
    unsigned short* __restrict__ zcl)
{
    __shared__ float tile[32][33];
    const int ht = blockIdx.x, ct = blockIdx.y, b = blockIdx.z;

    const float* src = z + ((size_t)(b * KDIM + ct * 32)) * HW + ht * 32;
    #pragma unroll
    for (int i = 0; i < 4; ++i) {
        int e  = threadIdx.x + i * 256;
        int cl = e >> 5, hl = e & 31;
        tile[cl][hl] = src[cl * HW + hl];
    }
    __syncthreads();
    unsigned short* dst = zcl + ((size_t)(b * HW + ht * 32)) * KDIM + ct * 32;
    #pragma unroll
    for (int i = 0; i < 4; ++i) {
        int e  = threadIdx.x + i * 256;
        int hl = e >> 5, cl = e & 31;
        dst[hl * KDIM + cl] = f32_to_bf16_rne(tile[cl][hl]);
    }
}

// ---------------------------------------------------------------------------
// Main: fused GEMM (d = ||e||^2 - 2 z.e) + argmin partials, bf16 WMMA.
// grid (64, NSPLIT) x 128 thr (4 waves). Each wave owns 64 M rows
// (4 A-fragment sets resident, 4 independent accumulator chains), so each
// staged B fragment feeds 4 WMMAs -> ds:wmma = 1:2. B tiles double-buffered
// in LDS via CDNA5 async global->LDS copies. ~19 KB LDS -> 2 blocks/WGP.
// ---------------------------------------------------------------------------
struct SMem {
    __align__(16) unsigned char Bt[2][BTILE];  // padded B tiles
    float cnS[512];                            // codebook norms slice
};

__global__ __launch_bounds__(128) void vq_argmin(
    const unsigned short* __restrict__ zcl,    // [NPOS, 256] bf16
    const unsigned short* __restrict__ cb16,   // [NE,   256] bf16
    const float* __restrict__ cnorm,           // [NE]
    float* __restrict__ pval,                  // [NPOS, NSPLIT]
    int*   __restrict__ pidx)                  // [NPOS, NSPLIT]
{
    __shared__ SMem smem;
    const unsigned int smemBtBase =
        (unsigned int)(unsigned long long)(void*)&smem.Bt[0][0];

    const int t     = threadIdx.x;               // 0..127
    const int wave  = t >> 5;                    // 0..3
    const int lane  = t & 31;
    const int ny    = blockIdx.y;
    const int nbase = ny * (NE / NSPLIT);        // 512-code slice
    const int mblk  = blockIdx.x * 256 + wave * 64;  // 64 rows per wave
    const int dA    = (lane < 16) ? 0 : 8;
    const int ncol  = lane & 15;
    const int hi    = (lane >> 4);               // 0 or 1

    // ---- stage cnorm slice ----
    for (int i = t; i < 512; i += 128)
        smem.cnS[i] = cnorm[nbase + i];

    // ---- issue async copy of B tile 0 into buffer 0 (512 x 16B chunks) ----
    {
        const unsigned char* g0 = (const unsigned char*)(cb16 + (size_t)nbase * KDIM);
        #pragma unroll
        for (int i = 0; i < 4; ++i) {
            int ch = t + i * 128;
            unsigned int la = smemBtBase + (unsigned)((ch >> 5) * BROW + (ch & 31) * 16);
            async_copy_b128(la, g0 + ch * 16);
        }
    }

    // ---- load 4 A fragment sets (rows mblk+16s .. +15, s=0..3) ----
    v16bf a[4][8];
    {
        const int r0 = mblk + (lane & 15);
        #pragma unroll
        for (int s = 0; s < 4; ++s) {
            #pragma unroll
            for (int c = 0; c < 8; ++c) {
                union { uint4 q[2]; v16bf v; } u;
                const unsigned short* base =
                    zcl + (size_t)(r0 + 16 * s) * KDIM + c * 32 + dA;
                u.q[0] = *(const uint4*)(base);
                u.q[1] = *(const uint4*)(base + 16);
                a[s][c] = u.v;
            }
        }
    }

    float minv[4][8];
    int   mini[4][8];
    #pragma unroll
    for (int s = 0; s < 4; ++s)
        #pragma unroll
        for (int r = 0; r < 8; ++r) { minv[s][r] = 3.402823466e38f; mini[s][r] = 0; }

    wait_asynccnt0();
    __syncthreads();

    const int fragOff = (lane & 15) * BROW + (lane & 16);  // row*528 + 0/16

    for (int nt = 0; nt < NT_PER; ++nt) {
        const int buf = nt & 1;

        // prefetch next tile into the other buffer
        if (nt + 1 < NT_PER) {
            const unsigned char* gn =
                (const unsigned char*)(cb16 + (size_t)(nbase + (nt + 1) * 16) * KDIM);
            #pragma unroll
            for (int i = 0; i < 4; ++i) {
                int ch = t + i * 128;
                unsigned int la = smemBtBase + (unsigned)((buf ^ 1) * BTILE +
                                   (ch >> 5) * BROW + (ch & 31) * 16);
                async_copy_b128(la, gn + ch * 16);
            }
        }

        // compute on current buffer: 32 WMMA (4 independent chains per B frag)
        const unsigned char* bbase = &smem.Bt[buf][0] + fragOff;
        v8f acc0 = {}, acc1 = {}, acc2 = {}, acc3 = {};
        #pragma unroll
        for (int c = 0; c < 8; ++c) {
            union { uint4 q[2]; v16bf v; } u;
            u.q[0] = *(const uint4*)(bbase + c * 64);
            u.q[1] = *(const uint4*)(bbase + c * 64 + 32);
            acc0 = __builtin_amdgcn_wmma_f32_16x16x32_bf16(
                       false, a[0][c], false, u.v, (short)0, acc0, false, false);
            acc1 = __builtin_amdgcn_wmma_f32_16x16x32_bf16(
                       false, a[1][c], false, u.v, (short)0, acc1, false, false);
            acc2 = __builtin_amdgcn_wmma_f32_16x16x32_bf16(
                       false, a[2][c], false, u.v, (short)0, acc2, false, false);
            acc3 = __builtin_amdgcn_wmma_f32_16x16x32_bf16(
                       false, a[3][c], false, u.v, (short)0, acc3, false, false);
        }

        const float cn = smem.cnS[nt * 16 + ncol];
        const int   n  = nbase + nt * 16 + ncol;
        #pragma unroll
        for (int r = 0; r < 8; ++r) {
            float d0 = fmaf(-2.0f, acc0[r], cn);
            float d1 = fmaf(-2.0f, acc1[r], cn);
            float d2 = fmaf(-2.0f, acc2[r], cn);
            float d3 = fmaf(-2.0f, acc3[r], cn);
            if (d0 < minv[0][r]) { minv[0][r] = d0; mini[0][r] = n; }
            if (d1 < minv[1][r]) { minv[1][r] = d1; mini[1][r] = n; }
            if (d2 < minv[2][r]) { minv[2][r] = d2; mini[2][r] = n; }
            if (d3 < minv[3][r]) { minv[3][r] = d3; mini[3][r] = n; }
        }

        wait_asynccnt0();
        __syncthreads();
    }

    // reduce min(+index) across the 16 lanes holding the same rows, then
    // write per-split partials. C layout: VGPR r -> row r (lanes 0-15),
    // row r+8 (lanes 16-31).
    #pragma unroll
    for (int s = 0; s < 4; ++s) {
        #pragma unroll
        for (int r = 0; r < 8; ++r) {
            float mv = minv[s][r]; int mi = mini[s][r];
            #pragma unroll
            for (int off = 1; off < 16; off <<= 1) {
                float ov = __shfl_xor(mv, off, 32);
                int   oi = __shfl_xor(mi, off, 32);
                if (ov < mv || (ov == mv && oi < mi)) { mv = ov; mi = oi; }
            }
            if ((lane & 15) == 0) {
                int row = mblk + 16 * s + r + (hi << 3);
                pval[(size_t)row * NSPLIT + ny] = mv;
                pidx[(size_t)row * NSPLIT + ny] = mi;
            }
        }
    }
}

// Merge per-split argmin partials (ties -> lowest index).
__global__ __launch_bounds__(256) void vq_merge(
    const float* __restrict__ pval,
    const int*   __restrict__ pidx,
    int*         __restrict__ index_out)
{
    int p = blockIdx.x * 256 + threadIdx.x;
    float best = pval[(size_t)p * NSPLIT];
    int   bi   = pidx[(size_t)p * NSPLIT];
    #pragma unroll
    for (int j = 1; j < NSPLIT; ++j) {
        float v = pval[(size_t)p * NSPLIT + j];
        int   i = pidx[(size_t)p * NSPLIT + j];
        if (v < best || (v == best && i < bi)) { best = v; bi = i; }
    }
    index_out[p] = bi;
}

// ---------------------------------------------------------------------------
// Gather z_q (f32 codebook), write channel-first coalesced, loss partials.
// grid (8 c-tiles, 16 batches) x 256 thr; each thread: 4 hw x 32 channels.
// ---------------------------------------------------------------------------
__global__ __launch_bounds__(256) void vq_gather_loss(
    const float* __restrict__ z,
    const float* __restrict__ cb_f32,
    const int* __restrict__ index,
    float* __restrict__ zq_out,
    float* __restrict__ partials)
{
    const int ct = blockIdx.x, b = blockIdx.y;
    const int t = threadIdx.x;

    float s = 0.0f;
    for (int hc = 0; hc < 4; ++hc) {
        const int hw  = hc * 256 + t;
        const int idx = index[b * HW + hw];
        #pragma unroll
        for (int cl = 0; cl < 32; ++cl) {
            const int c = ct * 32 + cl;
            float zq = cb_f32[(size_t)idx * KDIM + c];        // gather (L2)
            size_t o = ((size_t)(b * KDIM + c)) * HW + hw;    // coalesced
            float zv = z[o];
            zq_out[o] = zq;
            float d = zq - zv;
            s += d * d;
        }
    }

    #pragma unroll
    for (int off = 16; off > 0; off >>= 1)
        s += __shfl_xor(s, off, 32);
    __shared__ float red[8];
    if ((t & 31) == 0) red[t >> 5] = s;
    __syncthreads();
    if (t < 8) {
        float x = red[t];
        #pragma unroll
        for (int off = 4; off > 0; off >>= 1)
            x += __shfl_xor(x, off, 32);
        if (t == 0) partials[b * 8 + ct] = x;
    }
}

// Deterministic final loss reduction (no float atomics -> replay-stable).
__global__ __launch_bounds__(256) void vq_loss_reduce(
    const float* __restrict__ partials,
    float* __restrict__ loss_out)
{
    const int t = threadIdx.x;
    float s = (t < 128) ? partials[t] : 0.0f;
    #pragma unroll
    for (int off = 16; off > 0; off >>= 1)
        s += __shfl_xor(s, off, 32);
    __shared__ float red[8];
    if ((t & 31) == 0) red[t >> 5] = s;
    __syncthreads();
    if (t < 8) {
        float x = red[t];
        #pragma unroll
        for (int off = 4; off > 0; off >>= 1)
            x += __shfl_xor(x, off, 32);
        // loss = (1 + BETA) * mean(diff^2), BETA = 0.25, mean over 4194304
        if (t == 0) *loss_out = x * (1.25f / 4194304.0f);
    }
}

// ---------------------------------------------------------------------------
extern "C" void kernel_launch(void* const* d_in, const int* in_sizes, int n_in,
                              void* d_out, int out_size, void* d_ws, size_t ws_size,
                              hipStream_t stream) {
    const float* z   = (const float*)d_in[0];   // [16,256,32,32]
    const float* cbf = (const float*)d_in[1];   // [8192,256]

    float* out      = (float*)d_out;
    float* zq_out   = out;                           // 4194304 f32
    int*   idx_out  = (int*)(out + 4194304);         // 16384 i32 (raw bits)
    float* loss_out = out + 4194304 + 16384;         // 1 f32

    // workspace layout
    char* ws = (char*)d_ws;
    unsigned short* zcl   = (unsigned short*)(ws);                 //  8 MB
    unsigned short* cb16  = (unsigned short*)(ws + 8388608);       //  4 MB
    float*          cnorm = (float*)(ws + 12582912);               // 32 KB
    float*          pval  = (float*)(ws + 12615680);               //  1 MB
    int*            pidx  = (int*)  (ws + 13664256);               //  1 MB
    float*          parts = (float*)(ws + 14712832);               // 512 B

    vq_prep_codebook<<<NE / 8, 256, 0, stream>>>(cbf, cb16, cnorm);
    vq_prep_z<<<dim3(32, 8, 16), 256, 0, stream>>>(z, zcl);
    vq_argmin<<<dim3(NPOS / 256, NSPLIT), 128, 0, stream>>>(zcl, cb16, cnorm, pval, pidx);
    vq_merge<<<NPOS / 256, 256, 0, stream>>>(pval, pidx, idx_out);
    vq_gather_loss<<<dim3(8, 16), 256, 0, stream>>>(z, cbf, idx_out, zq_out, parts);
    vq_loss_reduce<<<1, 256, 0, stream>>>(parts, loss_out);
}
